// LocalSensitiveAttention_4672924418314
// MI455X (gfx1250) — compile-verified, matrix operand order checked by reading, not run on hardware
//
#include <hip/hip_runtime.h>
#include <math.h>

#define BB 64
#define TT 2048
#define D_DEC 1024
#define D_ENC 512
#define D_ATT 128
#define N_FILT 32
#define KCONV 31
#define LDSPITCH 40   // bf16 elements per tile row (80B stride -> conflict-free b128 loads)

typedef __attribute__((ext_vector_type(16))) __bf16 v16bf;
typedef __attribute__((ext_vector_type(8)))  float  v8f;

// Load a 16-element bf16 fragment (ISA 16-bit A/B layout) as two b128 LDS reads.
__device__ __forceinline__ v16bf load_frag(const __bf16* base, int row, int halfSel) {
    union { uint4 q[2]; v16bf v; } u;
    const uint4* p = (const uint4*)base;
    const int o = row * (LDSPITCH / 8) + (halfSel >> 3);
    u.q[0] = p[o];
    u.q[1] = p[o + 2];
    return u.v;
}

// branch-free tanh: exact limits at +-inf, ~5 VALU ops via v_exp_f32
__device__ __forceinline__ float fast_tanh(float x) {
    return 1.f - 2.f / (__expf(2.f * x) + 1.f);
}

// Per-thread staging registers: 16 enc floats (2 rows x 8) + 16 Wm bf16 (2 rows x 8)
struct StageRegs {
    float4 a0lo, a0hi, a1lo, a1hi;
    uint4  b0, b1;
};

__device__ __forceinline__ void stage_load(StageRegs& s, const float* __restrict__ enc,
                                           const __bf16* __restrict__ WmBf,
                                           int g0, int k0, int tid) {
    const int j0 = tid, j1 = tid + 256;
    {
        const int r = j0 >> 2, c = (j0 & 3) * 8;
        const float* p = enc + (size_t)(g0 + r) * D_ENC + (k0 + c);
        s.a0lo = *(const float4*)p; s.a0hi = *(const float4*)(p + 4);
    }
    {
        const int r = j1 >> 2, c = (j1 & 3) * 8;
        const float* p = enc + (size_t)(g0 + r) * D_ENC + (k0 + c);
        s.a1lo = *(const float4*)p; s.a1hi = *(const float4*)(p + 4);
    }
    {
        const int n = j0 >> 2, kc = (j0 & 3) * 8;
        s.b0 = *(const uint4*)&WmBf[(size_t)n * D_ENC + (k0 + kc)];
    }
    {
        const int n = j1 >> 2, kc = (j1 & 3) * 8;
        s.b1 = *(const uint4*)&WmBf[(size_t)n * D_ENC + (k0 + kc)];
    }
}

__device__ __forceinline__ uint4 pack8(const float4& lo, const float4& hi) {
    union { __bf16 bl[8]; uint4 u; } p;
    p.bl[0] = (__bf16)lo.x; p.bl[1] = (__bf16)lo.y;
    p.bl[2] = (__bf16)lo.z; p.bl[3] = (__bf16)lo.w;
    p.bl[4] = (__bf16)hi.x; p.bl[5] = (__bf16)hi.y;
    p.bl[6] = (__bf16)hi.z; p.bl[7] = (__bf16)hi.w;
    return p.u;
}

__device__ __forceinline__ void stage_store(const StageRegs& s, __bf16* As, __bf16* Bs,
                                            int tid) {
    const int j0 = tid, j1 = tid + 256;
    { const int r = j0 >> 2, c = (j0 & 3) * 8; *(uint4*)&As[r * LDSPITCH + c] = pack8(s.a0lo, s.a0hi); }
    { const int r = j1 >> 2, c = (j1 & 3) * 8; *(uint4*)&As[r * LDSPITCH + c] = pack8(s.a1lo, s.a1hi); }
    { const int n = j0 >> 2, kc = (j0 & 3) * 8; *(uint4*)&Bs[n * LDSPITCH + kc] = s.b0; }
    { const int n = j1 >> 2, kc = (j1 & 3) * 8; *(uint4*)&Bs[n * LDSPITCH + kc] = s.b1; }
}

// ---------------- Kernel 1: fold conv weights through Wloc ----------------
__global__ void __launch_bounds__(128)
prep_loc_kernel(const float* __restrict__ Wloc, const float* __restrict__ conv_w,
                const float* __restrict__ conv_b, float* __restrict__ Wcomb,
                float* __restrict__ locbias) {
    const int d = threadIdx.x;
    float lb = 0.f;
    for (int c = 0; c < N_FILT; ++c) lb += Wloc[d * N_FILT + c] * conv_b[c];
    locbias[d] = lb;
    for (int k = 0; k < KCONV; ++k) {
        float s = 0.f;
        for (int c = 0; c < N_FILT; ++c) s += Wloc[d * N_FILT + c] * conv_w[c * KCONV + k];
        Wcomb[d * KCONV + k] = s;
    }
}

// ---------------- Kernel 2: Wm -> bf16 ----------------
__global__ void __launch_bounds__(256)
wmcvt_kernel(const float* __restrict__ Wm, __bf16* __restrict__ WmBf) {
    const int n = blockIdx.x;
    for (int k = threadIdx.x; k < D_ENC; k += 256)
        WmBf[(size_t)n * D_ENC + k] = (__bf16)Wm[(size_t)n * D_ENC + k];
}

// ---------------- Kernel 3: q projection, bias + locbias folded in ----------------
__global__ void __launch_bounds__(128)
qproj_kernel(const float* __restrict__ hidden, const float* __restrict__ Wq,
             const float* __restrict__ bias, const float* __restrict__ locbias,
             float* __restrict__ qb) {
    const int b = blockIdx.x, d = threadIdx.x;
    const float* h = hidden + (size_t)b * D_DEC;
    const float* w = Wq + (size_t)d * D_DEC;
    float s = 0.f;
    for (int k = 0; k < D_DEC; ++k) s += h[k] * w[k];
    qb[b * D_ATT + d] = s + bias[d] + locbias[d];
}

// ---------------- Kernel 4: main WMMA kernel ----------------
__global__ void __launch_bounds__(256)
attn_main_kernel(const float* __restrict__ enc, const __bf16* __restrict__ WmBf,
                 const float* __restrict__ prev, const float* __restrict__ qb,
                 const float* __restrict__ Wcomb, const float* __restrict__ We,
                 float* __restrict__ mOut, float* __restrict__ eOut) {
    __shared__ __align__(16) __bf16 As[128 * LDSPITCH];
    __shared__ __align__(16) __bf16 Bs[128 * LDSPITCH];
    __shared__ float prevS[160];
    __shared__ float qbS[128];
    __shared__ float WeS[128];

    const int tid  = threadIdx.x;
    const int wave = tid >> 5;
    const int lane = tid & 31;
    const int col0 = lane & 15;
    const int half = (lane < 16) ? 0 : 8;
    const int g0   = blockIdx.x * 128;
    const int b    = g0 / TT;
    const int t0   = g0 % TT;
    const int rA   = wave * 16 + col0;

    if (tid < 128) { qbS[tid] = qb[b * D_ATT + tid]; WeS[tid] = We[tid]; }
    for (int i = tid; i < 160; i += 256) {
        const int t = t0 - 15 + i;
        prevS[i] = (t >= 0 && t < TT) ? prev[(size_t)b * TT + t] : 0.f;
    }

    v8f acc[8];
    const v8f vzero = {0.f, 0.f, 0.f, 0.f, 0.f, 0.f, 0.f, 0.f};
#pragma unroll
    for (int nt = 0; nt < 8; ++nt) acc[nt] = vzero;

    StageRegs cur, nxt;
    stage_load(cur, enc, WmBf, g0, 0, tid);

    for (int ks = 0; ks < 16; ++ks) {
        __syncthreads();                       // previous compute done: LDS writable
        stage_store(cur, As, Bs, tid);
        if (ks < 15) stage_load(nxt, enc, WmBf, g0, (ks + 1) * 32, tid);  // overlap HBM
        __syncthreads();                       // tiles ready

        const v16bf a = load_frag(As, rA, half);
        v16bf bcur = load_frag(Bs, col0, half);
#pragma unroll
        for (int nt = 0; nt < 8; ++nt) {
            v16bf bnxt;
            if (nt < 7) bnxt = load_frag(Bs, (nt + 1) * 16 + col0, half);  // prefetch
            acc[nt] = __builtin_amdgcn_wmma_f32_16x16x32_bf16(
                false, a, false, bcur, (short)0, acc[nt], false, false);
            bcur = bnxt;
        }
        cur = nxt;
    }

    // Spill m = enc @ Wm^T before the loc round.
#pragma unroll
    for (int nt = 0; nt < 8; ++nt) {
        const int n = nt * 16 + col0;
#pragma unroll
        for (int v = 0; v < 8; ++v) {
            const int r = wave * 16 + v + half;
            mOut[(size_t)(g0 + r) * D_ATT + n] = acc[nt][v];
        }
    }

    // loc round: A = Toeplitz(prev) 128x32 (k=31 zero), B = Wcomb^T bf16.
    __syncthreads();
    {
        const int i0 = tid * 16;               // 16 elements per thread, 4096 total
        const int r = i0 >> 5;
#pragma unroll
        for (int kk = 0; kk < 16; ++kk) {
            const int k = (i0 & 31) + kk;
            As[r * LDSPITCH + k] = (k < KCONV) ? (__bf16)prevS[r + k] : (__bf16)0.f;
        }
        const int n = r;
#pragma unroll
        for (int kk = 0; kk < 16; ++kk) {
            const int k = (i0 & 31) + kk;
            Bs[n * LDSPITCH + k] = (k < KCONV) ? (__bf16)Wcomb[n * KCONV + k] : (__bf16)0.f;
        }
    }
    __syncthreads();
    {
        const v16bf a = load_frag(As, rA, half);
        v16bf bcur = load_frag(Bs, col0, half);
#pragma unroll
        for (int nt = 0; nt < 8; ++nt) {
            v16bf bnxt;
            if (nt < 7) bnxt = load_frag(Bs, (nt + 1) * 16 + col0, half);
            acc[nt] = __builtin_amdgcn_wmma_f32_16x16x32_bf16(
                false, a, false, bcur, (short)0, acc[nt], false, false);
            bcur = bnxt;
        }
    }

    // Epilogue: x = tanh(q + m + loc), e = x . We
    float eacc[8];
#pragma unroll
    for (int v = 0; v < 8; ++v) eacc[v] = 0.f;
#pragma unroll
    for (int nt = 0; nt < 8; ++nt) {
        const int n = nt * 16 + col0;
        const float wE = WeS[n];
        const float qv = qbS[n];
#pragma unroll
        for (int v = 0; v < 8; ++v)
            eacc[v] += fast_tanh(qv + acc[nt][v]) * wE;
    }
#pragma unroll
    for (int v = 0; v < 8; ++v) {
        float s = eacc[v];
        for (int msk = 1; msk < 16; msk <<= 1) s += __shfl_xor(s, msk, 32);
        if (col0 == 0) {
            const int r = wave * 16 + v + half;
            eOut[(size_t)b * TT + (t0 + r)] = s;
        }
    }
}

// ---------------- Kernel 5: softmax over T per batch ----------------
__global__ void __launch_bounds__(256)
softmax_kernel(const float* __restrict__ eIn, float* __restrict__ attnOut) {
    __shared__ float red[256];
    const int b = blockIdx.x, tid = threadIdx.x;
    const float* e = eIn + (size_t)b * TT;
    float vals[8];
    float lm = -INFINITY;
#pragma unroll
    for (int i = 0; i < 8; ++i) { vals[i] = e[tid + i * 256]; lm = fmaxf(lm, vals[i]); }
    red[tid] = lm;
    __syncthreads();
    for (int s = 128; s > 0; s >>= 1) {
        if (tid < s) red[tid] = fmaxf(red[tid], red[tid + s]);
        __syncthreads();
    }
    const float mx = red[0];
    __syncthreads();
    float ls = 0.f;
#pragma unroll
    for (int i = 0; i < 8; ++i) { vals[i] = __expf(vals[i] - mx); ls += vals[i]; }
    red[tid] = ls;
    __syncthreads();
    for (int s = 128; s > 0; s >>= 1) {
        if (tid < s) red[tid] += red[tid + s];
        __syncthreads();
    }
    const float inv = 1.f / red[0];
#pragma unroll
    for (int i = 0; i < 8; ++i) attnOut[(size_t)b * TT + tid + i * 256] = vals[i] * inv;
}

// ---------------- Kernel 6: ctx partials + reduce (deterministic) ----------
__global__ void __launch_bounds__(128)
ctx_partial_kernel(const float* __restrict__ attn, const float* __restrict__ mIn,
                   float* __restrict__ pBuf) {
    const int b = blockIdx.x, ch = blockIdx.y, d = threadIdx.x;
    const int tBase = ch * (TT / 16);
    const float* mb = mIn + ((size_t)b * TT + tBase) * D_ATT;
    const float* ab = attn + (size_t)b * TT + tBase;
    float s = 0.f;
    for (int t = 0; t < TT / 16; ++t) s += ab[t] * mb[(size_t)t * D_ATT + d];
    pBuf[((size_t)b * 16 + ch) * D_ATT + d] = s;
}

__global__ void __launch_bounds__(128)
ctx_reduce_kernel(const float* __restrict__ pBuf, float* __restrict__ ctxOut) {
    const int b = blockIdx.x, d = threadIdx.x;
    float s = 0.f;
    for (int c = 0; c < 16; ++c) s += pBuf[((size_t)b * 16 + c) * D_ATT + d];
    ctxOut[b * D_ATT + d] = s;
}

extern "C" void kernel_launch(void* const* d_in, const int* in_sizes, int n_in,
                              void* d_out, int out_size, void* d_ws, size_t ws_size,
                              hipStream_t stream) {
    (void)in_sizes; (void)n_in; (void)out_size; (void)ws_size;
    const float* hidden = (const float*)d_in[0];
    const float* enc    = (const float*)d_in[1];
    const float* prev   = (const float*)d_in[2];
    const float* Wq     = (const float*)d_in[3];
    const float* Wm     = (const float*)d_in[4];
    const float* Wloc   = (const float*)d_in[5];
    const float* We     = (const float*)d_in[6];
    const float* conv_w = (const float*)d_in[7];
    const float* conv_b = (const float*)d_in[8];
    const float* bias   = (const float*)d_in[9];

    float* out     = (float*)d_out;
    float* ctxOut  = out;                          // [B,128]
    float* attnOut = out + BB * D_ATT;             // [B,T]

    char*   ws      = (char*)d_ws;
    float*  Wcomb   = (float*)(ws);                           // 128*31 f32
    float*  locbias = (float*)(ws + 16384);                   // 128 f32
    float*  qb      = (float*)(ws + 20480);                   // 64*128 f32
    __bf16* WmBf    = (__bf16*)(ws + 65536);                  // 128*512 bf16
    float*  pBuf    = (float*)(ws + 196608);                  // 64*16*128 f32
    float*  eBuf    = (float*)(ws + 786432);                  // B*T f32
    float*  mBuf    = (float*)(ws + (size_t)(2 << 20));       // B*T*128 f32 (64 MB)

    prep_loc_kernel<<<1, 128, 0, stream>>>(Wloc, conv_w, conv_b, Wcomb, locbias);
    wmcvt_kernel<<<D_ATT, 256, 0, stream>>>(Wm, WmBf);
    qproj_kernel<<<BB, D_ATT, 0, stream>>>(hidden, Wq, bias, locbias, qb);
    attn_main_kernel<<<(BB * TT) / 128, 256, 0, stream>>>(enc, WmBf, prev, qb, Wcomb, We,
                                                          mBuf, eBuf);
    softmax_kernel<<<BB, 256, 0, stream>>>(eBuf, attnOut);
    ctx_partial_kernel<<<dim3(BB, 16), 128, 0, stream>>>(attnOut, mBuf, pBuf);
    ctx_reduce_kernel<<<BB, 128, 0, stream>>>(pBuf, ctxOut);
}
